// TransNet_24584392802716
// MI455X (gfx1250) — compile-verified, matrix operand order checked by reading, not run on hardware
//
#include <hip/hip_runtime.h>

typedef unsigned short u16;
typedef __attribute__((ext_vector_type(16))) __bf16 v16bf;
typedef __attribute__((ext_vector_type(8)))  float  v8f;

#define NP 17
#define NB 16
#define OUT 256
#define BATCH 4096

__device__ __forceinline__ u16 f2bf(float f) {
    unsigned u = __float_as_uint(f);
    u += 0x7FFFu + ((u >> 16) & 1u);           // round-to-nearest-even
    return (u16)(u >> 16);
}
__device__ __forceinline__ float bf2f(u16 v) {
    return __uint_as_float(((unsigned)v) << 16);
}

union ABfrag { v16bf v; uint4 q[2]; };

// ---------------------------------------------------------------------------
// Grouped GEMM, bf16 WMMA, f32 accumulate.
//   A : [4096, G, K] bf16 (contiguous)        (activations)
//   Wp: packed weights [G][K/32][N/16][32 lanes][16 halves] bf16
//   out = act(A[:,g,:] @ W[g] + bias[g])  -> optional f32 and/or bf16 outputs
//         written at row stride ld*, column offset off* (for fused concat).
// Block: 256 thr = 8 waves, tile 128(M) x 256(N); wave tile 32x128.
// ---------------------------------------------------------------------------
__global__ void __launch_bounds__(256)
kGemmWmma(const u16* __restrict__ A, const u16* __restrict__ Wp,
          const float* __restrict__ bias,
          float* __restrict__ outF, int ldF, int offF,
          u16*  __restrict__ outB, int ldB, int offB,
          int G, int K, int N, int relu)
{
    const int lane = threadIdx.x & 31;
    const int wid  = threadIdx.x >> 5;
    const int g    = blockIdx.z;
    const int mBlock = blockIdx.x * 128 + (wid >> 1) * 32;
    const int nBlock = (wid & 1) * 128;

    const size_t ldA = (size_t)G * K;
    // A fragment base pointers (ISA 16-bit A layout: lanes>=16 hold K+8 block)
    const int khi = (lane >> 4) << 3;
    const u16* pa0 = A + (size_t)(mBlock + (lane & 15)) * ldA + (size_t)g * K + khi;
    const u16* pa1 = pa0 + 16 * ldA;

    const int KT = K >> 5;              // k-tiles of 32
    const int NT = N >> 4;              // n-tiles of 16
    const int nt0 = nBlock >> 4;

    v8f acc[2][8];
    #pragma unroll
    for (int mi = 0; mi < 2; ++mi)
        #pragma unroll
        for (int ni = 0; ni < 8; ++ni)
            acc[mi][ni] = (v8f){0,0,0,0,0,0,0,0};

    for (int kt = 0; kt < KT; ++kt) {
        const int kk = kt << 5;
        ABfrag a0, a1;
        a0.q[0] = *(const uint4*)(pa0 + kk);        // K = kk+khi   .. +7
        a0.q[1] = *(const uint4*)(pa0 + kk + 16);   // K = kk+16+khi.. +7
        a1.q[0] = *(const uint4*)(pa1 + kk);
        a1.q[1] = *(const uint4*)(pa1 + kk + 16);
        const u16* wrow = Wp + ((size_t)(g * KT + kt) * NT + nt0) * 512 + lane * 16;
        #pragma unroll
        for (int ni = 0; ni < 8; ++ni) {
            ABfrag b;
            b.q[0] = *(const uint4*)(wrow + (size_t)ni * 512);
            b.q[1] = *(const uint4*)(wrow + (size_t)ni * 512 + 8);
            acc[0][ni] = __builtin_amdgcn_wmma_f32_16x16x32_bf16(
                false, a0.v, false, b.v, (short)0, acc[0][ni], false, false);
            acc[1][ni] = __builtin_amdgcn_wmma_f32_16x16x32_bf16(
                false, a1.v, false, b.v, (short)0, acc[1][ni], false, false);
        }
    }

    // Epilogue: C/D f32 layout: lane l, reg r -> (M = r + 8*(l>=16), N = l&15)
    const int nc = lane & 15;
    const int mr = (lane >> 4) << 3;
    #pragma unroll
    for (int mi = 0; mi < 2; ++mi) {
        #pragma unroll
        for (int ni = 0; ni < 8; ++ni) {
            const int col = nBlock + ni * 16 + nc;
            const float bv = bias[(size_t)g * N + col];
            #pragma unroll
            for (int r = 0; r < 8; ++r) {
                float v = acc[mi][ni][r] + bv;
                if (relu) v = fmaxf(v, 0.0f);
                const size_t rowbase = (size_t)(mBlock + mi * 16 + mr + r) * G + g;
                if (outF) outF[rowbase * ldF + offF + col] = v;
                if (outB) outB[rowbase * ldB + offB + col] = f2bf(v);
            }
        }
    }
}

// ---------------------------------------------------------------------------
// Pack fp32 weights [G,K,N] into bf16 WMMA-B fragment tiles:
// packed[g][kt][nt][lane][h]  ==  W[g, kt*32 + kmap(lane,h), nt*16 + (lane&15)]
// kmap matches the ISA 16-bit operand layout used for A fragments.
// ---------------------------------------------------------------------------
__global__ void kPackW(const float* __restrict__ W, u16* __restrict__ out,
                       int K, int N, long total)
{
    long t = (long)blockIdx.x * blockDim.x + threadIdx.x;
    if (t >= total) return;
    const int h    = (int)(t & 15);
    const int lane = (int)((t >> 4) & 31);
    long rest = t >> 9;
    const int NT = N >> 4, KT = K >> 5;
    const int nt = (int)(rest % NT); rest /= NT;
    const int kt = (int)(rest % KT);
    const int g  = (int)(rest / KT);
    const int kb = (lane >= 16) ? 8 : 0;
    const int k  = (kt << 5) + ((h < 8) ? (kb + h) : (16 + kb + (h - 8)));
    const int n  = (nt << 4) + (lane & 15);
    out[t] = f2bf(W[((size_t)g * K + k) * N + n]);
}

// c1 first matmul: K=2, pure VALU. out bf16 [B,17,256]
__global__ void kC1(const float* __restrict__ x, const float* __restrict__ w1,
                    const float* __restrict__ b1, u16* __restrict__ out, long total)
{
    long t = (long)blockIdx.x * blockDim.x + threadIdx.x;
    if (t >= total) return;
    const int  o  = (int)(t & 255);
    const long bj = t >> 8;
    const int  j  = (int)(bj % NP);
    const float* xp = x + bj * 2;
    float v = xp[0] * w1[(j * 2 + 0) * OUT + o]
            + xp[1] * w1[(j * 2 + 1) * OUT + o]
            + b1[j * OUT + o];
    out[t] = f2bf(fmaxf(v, 0.0f));
}

// Adjacency aggregation: dst[b,j,c] = sum_k aff[b,j,k] * src[b,k,c]  (bf16 out)
// One block per batch sample; src slab staged in LDS.
__global__ void __launch_bounds__(256)
kAgg(const float* __restrict__ src, int ldSrc, const float* __restrict__ aff,
     u16* __restrict__ dst, int G, int C)
{
    __shared__ float sm[NP * 512 + NP * NP];
    float* sh = sm;
    float* sa = sm + G * C;
    const int b = blockIdx.x, tid = threadIdx.x;
    const size_t base = (size_t)b * G;
    for (int i = tid; i < G * C; i += 256) {
        int j = i / C, c = i - j * C;
        sh[i] = src[(base + j) * ldSrc + c];
    }
    for (int i = tid; i < G * G; i += 256) sa[i] = aff[(size_t)b * G * G + i];
    __syncthreads();
    for (int j = 0; j < G; ++j) {
        for (int c = tid; c < C; c += 256) {
            float a = 0.0f;
            for (int k = 0; k < G; ++k) a += sa[j * G + k] * sh[k * C + c];
            dst[(base + j) * C + c] = f2bf(a);
        }
    }
}

// joints -> bones: xb2[b,n,c] = sum_j inc[n,j] * h1[b,j,c]   (h1 = x2 cols 0..255)
__global__ void __launch_bounds__(256)
kInc(const float* __restrict__ x2, const float* __restrict__ inc,
     float* __restrict__ xb2)
{
    __shared__ float sh[NP * OUT];
    __shared__ float si[NB * NP];
    const int b = blockIdx.x, tid = threadIdx.x;
    for (int i = tid; i < NP * OUT; i += 256) {
        int j = i >> 8, c = i & 255;
        sh[i] = x2[((size_t)b * NP + j) * 512 + c];
    }
    for (int i = tid; i < NB * NP; i += 256) si[i] = inc[i];
    __syncthreads();
    for (int n = 0; n < NB; ++n) {
        for (int c = tid; c < OUT; c += 256) {
            float a = 0.0f;
            for (int j = 0; j < NP; ++j) a += si[n * NP + j] * sh[j * OUT + c];
            xb2[((size_t)b * NB + n) * 512 + c] = a;
        }
    }
}

// bones -> joints: X6b[b,j,768+c] = bf16( sum_n inc_inv[j,n] * XB6b[b,n,c] ), c<768
__global__ void __launch_bounds__(256)
kIncInv(const u16* __restrict__ xb6, const float* __restrict__ inv,
        u16* __restrict__ x6b)
{
    __shared__ float sh[NB * 768];
    __shared__ float si[NP * NB];
    const int b = blockIdx.x, tid = threadIdx.x;
    for (int i = tid; i < NB * 768; i += 256) sh[i] = bf2f(xb6[(size_t)b * NB * 768 + i]);
    for (int i = tid; i < NP * NB; i += 256) si[i] = inv[i];
    __syncthreads();
    for (int j = 0; j < NP; ++j) {
        for (int c = tid; c < 768; c += 256) {
            float a = 0.0f;
            for (int n = 0; n < NB; ++n) a += si[j * NB + n] * sh[n * 768 + c];
            x6b[((size_t)b * NP + j) * 1536 + 768 + c] = f2bf(a);
        }
    }
}

// f32 [rows, cols] -> bf16 at row stride ldDst (fused convert for concat)
__global__ void kCopyF2B(const float* __restrict__ src, u16* __restrict__ dst,
                         int cols, int ldDst, long total)
{
    long t = (long)blockIdx.x * blockDim.x + threadIdx.x;
    if (t >= total) return;
    long r = t / cols;
    int  c = (int)(t - r * cols);
    dst[r * (size_t)ldDst + c] = f2bf(src[t]);
}

// c9 head: per (b,j): t = h8 @ w1 + b1  (3 outs, K=256), out = t @ w2 + b2.
// One wave32 per (b,j); lane-strided K then shuffle reduce.
__global__ void __launch_bounds__(256)
kC9(const float* __restrict__ h8, const float* __restrict__ w1,
    const float* __restrict__ b1, const float* __restrict__ w2,
    const float* __restrict__ b2, float* __restrict__ out)
{
    const int lane = threadIdx.x & 31;
    const int warp = blockIdx.x * (blockDim.x >> 5) + (threadIdx.x >> 5);
    if (warp >= BATCH * NP) return;
    const int j = warp % NP;
    const float* hp = h8 + (size_t)warp * OUT;
    const float* wp = w1 + (size_t)j * OUT * 3;
    float a0 = 0.0f, a1 = 0.0f, a2 = 0.0f;
    for (int k = lane; k < OUT; k += 32) {
        float h = hp[k];
        a0 += h * wp[k * 3 + 0];
        a1 += h * wp[k * 3 + 1];
        a2 += h * wp[k * 3 + 2];
    }
    for (int off = 16; off; off >>= 1) {
        a0 += __shfl_xor(a0, off, 32);
        a1 += __shfl_xor(a1, off, 32);
        a2 += __shfl_xor(a2, off, 32);
    }
    if (lane == 0) {
        a0 += b1[j * 3 + 0]; a1 += b1[j * 3 + 1]; a2 += b1[j * 3 + 2];
        const float* w2p = w2 + j * 9;
        float* op = out + (size_t)warp * 3;
        op[0] = a0 * w2p[0] + a1 * w2p[3] + a2 * w2p[6] + b2[j * 3 + 0];
        op[1] = a0 * w2p[1] + a1 * w2p[4] + a2 * w2p[7] + b2[j * 3 + 1];
        op[2] = a0 * w2p[2] + a1 * w2p[5] + a2 * w2p[8] + b2[j * 3 + 2];
    }
}

// ---------------------------------------------------------------------------
extern "C" void kernel_launch(void* const* d_in, const int* in_sizes, int n_in,
                              void* d_out, int out_size, void* d_ws, size_t ws_size,
                              hipStream_t stream)
{
    (void)in_sizes; (void)n_in; (void)out_size; (void)ws_size;
    const float* x        = (const float*)d_in[0];
    const float* aff      = (const float*)d_in[1];
    const float* aff_bone = (const float*)d_in[2];
    const float* inc      = (const float*)d_in[3];
    const float* inc_inv  = (const float*)d_in[4];
    const float* W[33]; for (int i = 5; i < 33; ++i) W[i] = (const float*)d_in[i];

    // ---- workspace carve-up (256B aligned) ----
    char* wsp = (char*)d_ws;
    size_t off = 0;
    auto alloc = [&](size_t bytes) -> void* {
        size_t o = (off + 255) & ~(size_t)255;
        off = o + bytes;
        return (void*)(wsp + o);
    };
    u16* c1w2p = (u16*)alloc((size_t)NP * 256 * 256 * 2);
    u16* c2w1p = (u16*)alloc((size_t)NP * 256 * 256 * 2);
    u16* c2w2p = (u16*)alloc((size_t)NP * 256 * 256 * 2);
    u16* c3w1p = (u16*)alloc((size_t)NP * 512 * 256 * 2);
    u16* c3w2p = (u16*)alloc((size_t)NP * 256 * 256 * 2);
    u16* b2w1p = (u16*)alloc((size_t)NB * 256 * 256 * 2);
    u16* b2w2p = (u16*)alloc((size_t)NB * 256 * 256 * 2);
    u16* b3w1p = (u16*)alloc((size_t)NB * 512 * 256 * 2);
    u16* b3w2p = (u16*)alloc((size_t)NB * 256 * 256 * 2);
    u16* c8w1p = (u16*)alloc((size_t)NP * 1536 * 256 * 2);
    u16* c8w2p = (u16*)alloc((size_t)NP * 256 * 256 * 2);

    u16*   bufA   = (u16*)alloc((size_t)BATCH * NP * 256 * 2);   // bf16 ping
    u16*   bufB   = (u16*)alloc((size_t)BATCH * NP * 256 * 2);   // bf16 pong
    u16*   buf512 = (u16*)alloc((size_t)BATCH * NP * 512 * 2);   // bf16 K=512 agg
    u16*   X6b    = (u16*)alloc((size_t)BATCH * NP * 1536 * 2);  // c8 input, bf16
    float* x2f    = (float*)alloc((size_t)BATCH * NP * 512 * 4); // [h1 | c2out]; later h8
    float* xb2f   = (float*)alloc((size_t)BATCH * NB * 512 * 4); // [featb | b2out]
    u16*   xb6b   = (u16*)alloc((size_t)BATCH * NB * 768 * 2);   // bone concat bf16
    float* h8     = x2f;  // x2f dead after copy into X6b; reuse for c8 output

    const dim3 blk(256);
    auto cdiv = [](long a, long b) { return (unsigned)((a + b - 1) / b); };

    auto pack = [&](const float* Wf, u16* o, int G, int K, int N) {
        long tot = (long)G * K * N;
        kPackW<<<cdiv(tot, 256), blk, 0, stream>>>(Wf, o, K, N, tot);
    };
    auto gemm = [&](const u16* A, const u16* Wp, const float* bias,
                    float* oF, int ldF, int ofF, u16* oB, int ldB, int ofB,
                    int G, int K, int relu) {
        dim3 grid(BATCH / 128, 1, G);
        kGemmWmma<<<grid, blk, 0, stream>>>(A, Wp, bias, oF, ldF, ofF,
                                            oB, ldB, ofB, G, K, 256, relu);
    };

    // ---- 0) pack all WMMA weights to bf16 fragment layout ----
    pack(W[7],  c1w2p, NP, 256, 256);
    pack(W[9],  c2w1p, NP, 256, 256);
    pack(W[11], c2w2p, NP, 256, 256);
    pack(W[13], c3w1p, NP, 512, 256);
    pack(W[15], c3w2p, NP, 256, 256);
    pack(W[17], b2w1p, NB, 256, 256);
    pack(W[19], b2w2p, NB, 256, 256);
    pack(W[21], b3w1p, NB, 512, 256);
    pack(W[23], b3w2p, NB, 256, 256);
    pack(W[25], c8w1p, NP, 1536, 256);
    pack(W[27], c8w2p, NP, 256, 256);

    // ---- 1) c1 ----
    {
        long tot = (long)BATCH * NP * 256;
        kC1<<<cdiv(tot, 256), blk, 0, stream>>>(x, W[5], W[6], bufA, tot);
    }
    gemm(bufA, c1w2p, W[8], x2f, 512, 0, nullptr, 0, 0, NP, 256, 1); // h1 -> x2[:,0:256]

    // ---- 2) c2 ----
    kAgg<<<BATCH, blk, 0, stream>>>(x2f, 512, aff, bufB, NP, 256);
    gemm(bufB, c2w1p, W[10], nullptr, 0, 0, bufA, 256, 0, NP, 256, 1);
    gemm(bufA, c2w2p, W[12], x2f, 512, 256, nullptr, 0, 0, NP, 256, 1); // x2[:,256:512]

    // ---- 3) c3 ----
    kAgg<<<BATCH, blk, 0, stream>>>(x2f, 512, aff, buf512, NP, 512);
    gemm(buf512, c3w1p, W[14], nullptr, 0, 0, bufA, 256, 0, NP, 512, 1);
    gemm(bufA, c3w2p, W[16], nullptr, 0, 0, X6b, 1536, 512, NP, 256, 1); // X6b[:,512:768]

    // ---- 4) bone branch ----
    kInc<<<BATCH, blk, 0, stream>>>(x2f, inc, xb2f);                 // xb2[:,0:256]
    kAgg<<<BATCH, blk, 0, stream>>>(xb2f, 512, aff_bone, bufB, NB, 256);
    gemm(bufB, b2w1p, W[18], nullptr, 0, 0, bufA, 256, 0, NB, 256, 1);
    gemm(bufA, b2w2p, W[20], xb2f, 512, 256, nullptr, 0, 0, NB, 256, 1); // xb2[:,256:512]
    kAgg<<<BATCH, blk, 0, stream>>>(xb2f, 512, aff_bone, buf512, NB, 512);
    gemm(buf512, b3w1p, W[22], nullptr, 0, 0, bufA, 256, 0, NB, 512, 1);
    gemm(bufA, b3w2p, W[24], nullptr, 0, 0, xb6b, 768, 512, NB, 256, 1); // xb6[:,512:768]
    {
        long tot = (long)BATCH * NB * 512;
        kCopyF2B<<<cdiv(tot, 256), blk, 0, stream>>>(xb2f, xb6b, 512, 768, tot);
    }
    kIncInv<<<BATCH, blk, 0, stream>>>(xb6b, inc_inv, X6b);          // X6b[:,768:1536]

    // ---- 5) assemble c8 input: X6b[:,0:512] = bf16(x2) ----
    {
        long tot = (long)BATCH * NP * 512;
        kCopyF2B<<<cdiv(tot, 256), blk, 0, stream>>>(x2f, X6b, 512, 1536, tot);
    }

    // ---- 6) c8 (x2f now dead; its space becomes h8) ----
    gemm(X6b, c8w1p, W[26], nullptr, 0, 0, bufA, 256, 0, NP, 1536, 1);
    gemm(bufA, c8w2p, W[28], h8, 256, 0, nullptr, 0, 0, NP, 256, 1);

    // ---- 7) c9 head ----
    {
        int warps = BATCH * NP;
        kC9<<<cdiv(warps, 8), blk, 0, stream>>>(h8, W[29], W[30], W[31], W[32],
                                                (float*)d_out);
    }
}
